// TriangularGatedSelfAttention_36060545417657
// MI455X (gfx1250) — matrix operand order repaired and checked by reading, not compiled, not run on hardware
//
#include <hip/hip_runtime.h>
#include <hip/hip_bf16.h>
#include <math.h>

typedef __attribute__((ext_vector_type(16))) _Float16 v16h;
typedef __attribute__((ext_vector_type(8)))  _Float16 v8h;
typedef __attribute__((ext_vector_type(8)))  float    v8f;
typedef __attribute__((ext_vector_type(4)))  float    v4f;

#define S_DIM 256
#define C_DIM 128
#define NHEAD 4
#define HID   32
#define ATT_SCALE 0.17677669529663687f  // 1/sqrt(32)

// ---- workspace layout (bytes), all 256B aligned ----
#define WS_WT    0u                       // 512*128 f16  = 131072
#define WS_WOT   131072u                  // 128*128 f16  = 32768
#define WS_BIAS  163840u                  // 4 head-planes of 65536 f32 = 1048576
#define WS_Q     1212416u                 // 4*256*256*32 f16 = 16777216
#define WS_K     (WS_Q  + 16777216u)
#define WS_VT    (WS_K  + 16777216u)
#define WS_G     (WS_VT + 16777216u)
#define WS_O     (WS_G  + 16777216u)      // end = 85098496 bytes (~85 MB)

// ---------------- WMMA fragment helpers (wave32, 16x16x32 f16) ----------------
// A fragment (16 rows x 32 K): lane = row (m=lane&15, half=lane>>4),
//   per lane two contiguous 16B chunks at K = half*8 and K = 16+half*8.
__device__ __forceinline__ v16h frag_a(const _Float16* base, int row0, int ld, int k0) {
  const int lane = threadIdx.x & 31;
  const int m = lane & 15, h = lane >> 4;
  const _Float16* p = base + (size_t)(row0 + m) * ld + k0 + h * 8;
  v8h lo = *(const v8h*)p;
  v8h hi = *(const v8h*)(p + 16);
  v16h f;
#pragma unroll
  for (int t = 0; t < 8; ++t) { f[t] = lo[t]; f[t + 8] = hi[t]; }
  return f;
}
// B fragment (32 K x 16 cols), loaded from an N-major (col-major) matrix:
//   lane = col n, per lane two contiguous 16B chunks at K = half*8 and K = 16+half*8.
__device__ __forceinline__ v16h frag_b(const _Float16* base, int n0, int ld, int k0) {
  const int lane = threadIdx.x & 31;
  const int n = lane & 15, h = lane >> 4;
  const _Float16* p = base + (size_t)(n0 + n) * ld + k0 + h * 8;
  v8h lo = *(const v8h*)p;
  v8h hi = *(const v8h*)(p + 16);
  v16h f;
#pragma unroll
  for (int t = 0; t < 8; ++t) { f[t] = lo[t]; f[t + 8] = hi[t]; }
  return f;
}
#define WMMA_F16(a, b, c) \
  __builtin_amdgcn_wmma_f32_16x16x32_f16(false, (a), false, (b), (short)0, (c), false, false)

// Async DMA: copy 16B from global memory directly into LDS (gfx1250,
// GLOBAL_LOAD_ASYNC_TO_LDS_B128, tracked by ASYNCcnt). Low 32 bits of a
// generic pointer to __shared__ are the LDS byte address (flat aperture rule).
#define ASYNC_CP16(lds_ptr, g_ptr)                                             \
  do {                                                                         \
    unsigned _l = (unsigned)(uintptr_t)(lds_ptr);                              \
    unsigned long long _g = (unsigned long long)(uintptr_t)(g_ptr);            \
    asm volatile("global_load_async_to_lds_b128 %0, %1, off"                   \
                 :: "v"(_l), "v"(_g) : "memory");                              \
  } while (0)
#define ASYNC_WAIT0() asm volatile("s_wait_asynccnt 0x0" ::: "memory")

// ---------------- kernel 0: weight transpose + f16 convert ----------------
__global__ __launch_bounds__(128) void k_prep(
    const float* __restrict__ Wq, const float* __restrict__ Wk,
    const float* __restrict__ Wv, const float* __restrict__ Wg,
    const float* __restrict__ Wo, _Float16* __restrict__ WT, _Float16* __restrict__ WoT) {
  const int n = blockIdx.x;    // 0..639
  const int c = threadIdx.x;   // 0..127
  if (n < 512) {
    const float* W = (n < 128) ? Wq : (n < 256) ? Wk : (n < 384) ? Wv : Wg;
    WT[(size_t)n * 128 + c] = (_Float16)W[(size_t)c * 128 + (n & 127)];
  } else {
    const int co = n - 512;
    WoT[(size_t)co * 128 + c] = (_Float16)Wo[(size_t)c * 128 + co];
  }
}

// ---------------- kernel 1: LayerNorm + QKVG/bias projections ----------------
__global__ __launch_bounds__(128) void k_ln_proj(
    const float* __restrict__ x, const float* __restrict__ gamma, const float* __restrict__ beta,
    const float* __restrict__ Wb, const float* __restrict__ bg,
    const _Float16* __restrict__ WT,
    _Float16* __restrict__ Qh, _Float16* __restrict__ Kh, _Float16* __restrict__ Vt,
    _Float16* __restrict__ Gh, float* __restrict__ biasP) {
  __shared__ _Float16 zt[64 * 136];          // 64 rows x 128ch (+8 pad) f16
  const int tid = threadIdx.x;
  const int wv = tid >> 5, lane = tid & 31;
  const int m0 = blockIdx.x * 64;

  // warm this wave's weight panel (wave wv uses WT rows [wv*128, wv*128+128))
  __builtin_prefetch(WT + ((size_t)wv * 128 + lane * 4) * 128, 0, 3);

  // ---- LayerNorm (one wave per row, 4 channels per lane) + bias projection
  for (int rr = wv; rr < 64; rr += 4) {
    const int pos = m0 + rr;
    v4f xv = *(const v4f*)(x + (size_t)pos * 128 + lane * 4);
    float s  = xv[0] + xv[1] + xv[2] + xv[3];
    float s2 = xv[0]*xv[0] + xv[1]*xv[1] + xv[2]*xv[2] + xv[3]*xv[3];
#pragma unroll
    for (int msk = 16; msk >= 1; msk >>= 1) {
      s  += __shfl_xor(s,  msk, 32);
      s2 += __shfl_xor(s2, msk, 32);
    }
    const float mu  = s * (1.0f / 128.0f);
    const float var = s2 * (1.0f / 128.0f) - mu * mu;
    const float rs  = rsqrtf(var + 1e-5f);
    float ba[4] = {0.f, 0.f, 0.f, 0.f};
#pragma unroll
    for (int t = 0; t < 4; ++t) {
      const int c = lane * 4 + t;
      const float z = (xv[t] - mu) * rs * gamma[c] + beta[c];
      zt[rr * 136 + c] = (_Float16)z;
#pragma unroll
      for (int n = 0; n < 4; ++n) ba[n] += z * Wb[c * 4 + n];
    }
#pragma unroll
    for (int n = 0; n < 4; ++n) {
#pragma unroll
      for (int msk = 16; msk >= 1; msk >>= 1) ba[n] += __shfl_xor(ba[n], msk, 32);
    }
    // per-head planes: biasP[head][i*256 + k] -> coalesced reads in k_attn
    if (lane < 4) biasP[(size_t)lane * 65536 + pos] = ba[lane];
  }
  __syncthreads();

  // ---- projection: wave wv owns output matrix wv (0=q,1=k,2=v,3=g)
  // nt outer / rt inner, both rolled: keeps only 4 B-fragments (32 VGPRs) live
  // so the kernel stays under 256 VGPRs (avoids s_set_vgpr_msb churn).
  const int h = lane >> 4, nl = lane & 15;
  for (int nt = 0; nt < 8; ++nt) {
    const int n0 = wv * 128 + nt * 16;
    const v16h b0 = frag_b(WT, n0, 128, 0);
    const v16h b1 = frag_b(WT, n0, 128, 32);
    const v16h b2 = frag_b(WT, n0, 128, 64);
    const v16h b3 = frag_b(WT, n0, 128, 96);
    const int col  = n0 + nl;
    const int head = (col >> 5) & 3;
    const int d    = col & 31;
    const float bgv = (wv == 3) ? bg[head * 32 + d] : 0.0f;   // hoisted per tile
    for (int rt = 0; rt < 4; ++rt) {
      v8f acc = {0.f, 0.f, 0.f, 0.f, 0.f, 0.f, 0.f, 0.f};
      acc = WMMA_F16(frag_a(zt, rt * 16, 136, 0),  b0, acc);
      acc = WMMA_F16(frag_a(zt, rt * 16, 136, 32), b1, acc);
      acc = WMMA_F16(frag_a(zt, rt * 16, 136, 64), b2, acc);
      acc = WMMA_F16(frag_a(zt, rt * 16, 136, 96), b3, acc);
#pragma unroll
      for (int r = 0; r < 8; ++r) {
        const int pos = m0 + rt * 16 + r + 8 * h;
        const int i = pos >> 8, j = pos & 255;
        const float v = acc[r];
        if (wv == 0) {            // Q, pre-scaled, [n][j][i][d]
          Qh[(((size_t)head * 256 + j) * 256 + i) * 32 + d] = (_Float16)(v * ATT_SCALE);
        } else if (wv == 1) {     // K, [n][j][k][d]
          Kh[(((size_t)head * 256 + j) * 256 + i) * 32 + d] = (_Float16)v;
        } else if (wv == 2) {     // V transposed, [n][j][d][k]
          Vt[(((size_t)head * 256 + j) * 32 + d) * 256 + i] = (_Float16)v;
        } else {                  // gate = sigmoid(z@Wg + bg), [n][j][i][d]
          const float gv = __builtin_amdgcn_rcpf(1.0f + __expf(-(v + bgv)));
          Gh[(((size_t)head * 256 + j) * 256 + i) * 32 + d] = (_Float16)gv;
        }
      }
    }
  }
}

// ---------------- kernel 2: per-(head, j) attention ----------------
__global__ __launch_bounds__(256) void k_attn(
    const _Float16* __restrict__ Qh, const _Float16* __restrict__ Kh,
    const _Float16* __restrict__ Vt, const _Float16* __restrict__ Gh,
    const float* __restrict__ biasP, _Float16* __restrict__ O) {
  __shared__ _Float16 Ks[256 * 40];      // K  [k][d], row stride 40 (pad)
  __shared__ _Float16 Vs[32 * 264];      // Vt [d][k], row stride 264 (pad)
  __shared__ _Float16 As[8][16 * 40];    // per-wave attn repack, stride 40
  const int tid = threadIdx.x;
  const int wv = tid >> 5, lane = tid & 31;
  const int head = blockIdx.x >> 8, j = blockIdx.x & 255;
  const _Float16* Kg = Kh + ((size_t)head * 256 + j) * 256 * 32;
  const _Float16* Vg = Vt + ((size_t)head * 256 + j) * 32 * 256;
  const _Float16* Qg = Qh + ((size_t)head * 256 + j) * 256 * 32;
  const _Float16* Gg = Gh + ((size_t)head * 256 + j) * 256 * 32;
  const float*    biasH = biasP + (size_t)head * 65536;

  // warm Q and gate streams while K/V async staging is in flight
  __builtin_prefetch(Qg + (size_t)tid * 32, 0, 3);   // 256 thr x 64B = 16KB
  __builtin_prefetch(Gg + (size_t)tid * 32, 0, 3);

  // K: 256 rows x 32 f16 = 1024 x 16B chunks; V: 32 rows x 256 f16 = 1024 chunks.
  // DMA straight into LDS (ASYNCcnt), no VGPR round-trip.
  for (int c = tid; c < 1024; c += 256) {
    const int row = c >> 2, part = c & 3;
    ASYNC_CP16(Ks + row * 40 + part * 8, Kg + row * 32 + part * 8);
  }
  for (int c = tid; c < 1024; c += 256) {
    const int row = c >> 5, part = c & 31;
    ASYNC_CP16(Vs + row * 264 + part * 8, Vg + row * 256 + part * 8);
  }
  ASYNC_WAIT0();
  __syncthreads();

  const int h = lane >> 4, nl = lane & 15;
  for (int it = wv; it < 16; it += 8) {        // 2 i-tiles per wave
    const int i0 = it * 16;
    const v16h aq = frag_a(Qg, i0, 32, 0);     // full d=32 in one A fragment
    float dt[16][8];
#pragma unroll
    for (int kt = 0; kt < 16; ++kt) {          // logits: 16 WMMAs cover k=0..255
      v8f z = {0.f, 0.f, 0.f, 0.f, 0.f, 0.f, 0.f, 0.f};
      v8f acc = WMMA_F16(aq, frag_b(Ks, kt * 16, 40, 0), z);
      const int k = kt * 16 + nl;
#pragma unroll
      for (int r = 0; r < 8; ++r) {
        const int i = i0 + r + 8 * h;
        dt[kt][r] = acc[r] + biasH[(size_t)i * 256 + k];   // coalesced 4B/lane
      }
    }
    // softmax over k: per-lane fold over 16 tiles, then 16-lane half-wave shuffle reduce
    float inv[8];
#pragma unroll
    for (int r = 0; r < 8; ++r) {
      float m = dt[0][r];
#pragma unroll
      for (int kt = 1; kt < 16; ++kt) m = fmaxf(m, dt[kt][r]);
#pragma unroll
      for (int msk = 8; msk >= 1; msk >>= 1) m = fmaxf(m, __shfl_xor(m, msk, 32));
      float s = 0.0f;
#pragma unroll
      for (int kt = 0; kt < 16; ++kt) {
        const float e = __expf(dt[kt][r] - m);
        dt[kt][r] = e;
        s += e;
      }
#pragma unroll
      for (int msk = 8; msk >= 1; msk >>= 1) s += __shfl_xor(s, msk, 32);
      inv[r] = __builtin_amdgcn_rcpf(s);       // v_rcp_f32, not IEEE div chain
    }
    // attn @ V: repack C-layout logits -> A-layout f16 via per-wave LDS tile
    v8f oc0 = {0.f, 0.f, 0.f, 0.f, 0.f, 0.f, 0.f, 0.f};
    v8f oc1 = {0.f, 0.f, 0.f, 0.f, 0.f, 0.f, 0.f, 0.f};
    _Float16* stg = &As[wv][0];
#pragma unroll
    for (int kc = 0; kc < 8; ++kc) {           // k in chunks of 32
#pragma unroll
      for (int kk = 0; kk < 2; ++kk) {
        const int kt = kc * 2 + kk;
#pragma unroll
        for (int r = 0; r < 8; ++r)
          stg[(r + 8 * h) * 40 + kk * 16 + nl] = (_Float16)(dt[kt][r] * inv[r]);
      }
      const v16h aa = frag_a(stg, 0, 40, 0);
      oc0 = WMMA_F16(aa, frag_b(Vs, 0,  264, kc * 32), oc0);
      oc1 = WMMA_F16(aa, frag_b(Vs, 16, 264, kc * 32), oc1);
    }
    // gate + store to [i][j][n*d] f16
#pragma unroll
    for (int r = 0; r < 8; ++r) {
      const int i = i0 + r + 8 * h;
      const float g0 = (float)Gg[(size_t)i * 32 + nl];
      const float g1 = (float)Gg[(size_t)i * 32 + 16 + nl];
      O[((size_t)i * 256 + j) * 128 + head * 32 + nl]      = (_Float16)(oc0[r] * g0);
      O[((size_t)i * 256 + j) * 128 + head * 32 + 16 + nl] = (_Float16)(oc1[r] * g1);
    }
  }
}

// ---------------- kernel 3: output projection O @ Wo + bo ----------------
__global__ __launch_bounds__(128) void k_out(
    const _Float16* __restrict__ O, const _Float16* __restrict__ WoT,
    const float* __restrict__ bo, float* __restrict__ out) {
  const int tid = threadIdx.x;
  const int wv = tid >> 5, lane = tid & 31;
  const int h = lane >> 4, nl = lane & 15;
  const int m0 = blockIdx.x * 64;
  for (int rt = 0; rt < 4; ++rt) {
    const int row0 = m0 + rt * 16;
    const _Float16* Ob = O + (size_t)row0 * 128;
    const v16h a0 = frag_a(Ob, 0, 128, 0);
    const v16h a1 = frag_a(Ob, 0, 128, 32);
    const v16h a2 = frag_a(Ob, 0, 128, 64);
    const v16h a3 = frag_a(Ob, 0, 128, 96);
#pragma unroll
    for (int ntt = 0; ntt < 2; ++ntt) {
      const int n0 = wv * 32 + ntt * 16;
      v8f acc = {0.f, 0.f, 0.f, 0.f, 0.f, 0.f, 0.f, 0.f};
      acc = WMMA_F16(a0, frag_b(WoT, n0, 128, 0),  acc);
      acc = WMMA_F16(a1, frag_b(WoT, n0, 128, 32), acc);
      acc = WMMA_F16(a2, frag_b(WoT, n0, 128, 64), acc);
      acc = WMMA_F16(a3, frag_b(WoT, n0, 128, 96), acc);
      const int col = n0 + nl;
      const float bov = bo[col];
#pragma unroll
      for (int r = 0; r < 8; ++r)
        out[(size_t)(row0 + r + 8 * h) * 128 + col] = acc[r] + bov;
    }
  }
}

extern "C" void kernel_launch(void* const* d_in, const int* in_sizes, int n_in,
                              void* d_out, int out_size, void* d_ws, size_t ws_size,
                              hipStream_t stream) {
  const float* pair  = (const float*)d_in[0];
  const float* gamma = (const float*)d_in[1];
  const float* beta  = (const float*)d_in[2];
  const float* Wq    = (const float*)d_in[3];
  const float* Wk    = (const float*)d_in[4];
  const float* Wv    = (const float*)d_in[5];
  const float* Wb    = (const float*)d_in[6];
  const float* Wg    = (const float*)d_in[7];
  const float* bg    = (const float*)d_in[8];
  const float* Wo    = (const float*)d_in[9];
  const float* bo    = (const float*)d_in[10];

  char* ws = (char*)d_ws;
  _Float16* WT    = (_Float16*)(ws + WS_WT);
  _Float16* WoT   = (_Float16*)(ws + WS_WOT);
  float*    biasP = (float*)   (ws + WS_BIAS);
  _Float16* Qh    = (_Float16*)(ws + WS_Q);
  _Float16* Kh    = (_Float16*)(ws + WS_K);
  _Float16* Vt    = (_Float16*)(ws + WS_VT);
  _Float16* Gh    = (_Float16*)(ws + WS_G);
  _Float16* O     = (_Float16*)(ws + WS_O);

  k_prep<<<640, 128, 0, stream>>>(Wq, Wk, Wv, Wg, Wo, WT, WoT);
  k_ln_proj<<<1024, 128, 0, stream>>>(pair, gamma, beta, Wb, bg, WT, Qh, Kh, Vt, Gh, biasP);
  k_attn<<<1024, 256, 0, stream>>>(Qh, Kh, Vt, Gh, biasP, O);
  k_out<<<1024, 128, 0, stream>>>(O, WoT, bo, (float*)d_out);
}
